// CayleyStringAttention_19851338842655
// MI455X (gfx1250) — matrix operand order repaired and checked by reading, not compiled, not run on hardware
//
#include <hip/hip_runtime.h>
#include <hip/hip_bf16.h>
#include <math.h>

// ---------------------------------------------------------------------------
// CayleyStringAttention for MI455X (gfx1250), wave32, WMMA bf16 (fp32 accum).
// Pre-converted bf16 operands, double-buffered LDS GEMM, fragment preloading
// so DS loads of fragment n+1 overlap WMMA n (graduated s_wait_dscnt).
// ---------------------------------------------------------------------------

#define HIDDEN   768
#define NHEADS   12
#define HDIM     64
#define BATCH    64
#define SEQ      197
#define NPATCH   196
#define BNTOK    (BATCH * SEQ)          // 12608
#define QKVCOLS  (3 * HIDDEN)           // 2304
#define EPSV     1e-5f

typedef __bf16 bf16_t;
typedef __attribute__((ext_vector_type(16))) __bf16 v16bf;
typedef __attribute__((ext_vector_type(8)))  float  v8f;

__device__ __forceinline__ bf16_t f2bf(float f) {
    unsigned u = __builtin_bit_cast(unsigned, f);
    unsigned r = u + 0x7FFFu + ((u >> 16) & 1u);     // round-to-nearest-even
    unsigned short h = (unsigned short)(r >> 16);
    return __builtin_bit_cast(bf16_t, h);
}

__device__ __forceinline__ unsigned pack2bf(float a, float b) {
    unsigned ua = __builtin_bit_cast(unsigned, a);
    unsigned ub = __builtin_bit_cast(unsigned, b);
    ua = ua + 0x7FFFu + ((ua >> 16) & 1u);
    ub = ub + 0x7FFFu + ((ub >> 16) & 1u);
    return (ua >> 16) | (ub & 0xFFFF0000u);
}

__device__ __forceinline__ v8f wmma_bf16(v16bf a, v16bf b, v8f c) {
    return __builtin_amdgcn_wmma_f32_16x16x32_bf16(
        false, a, false, b, (short)0, c, false, false);
}

// A operand: 16(M) x 32(K) bf16 tile, row-major, row stride ld.
// Lane l holds row (l&15); K split 8-wide interleaved per half-wave.
__device__ __forceinline__ v16bf load_frag_a(const bf16_t* __restrict__ base, int ld) {
    int lane = threadIdx.x & 31;
    int row  = lane & 15;
    int kb   = (lane >> 4) << 3;
    const bf16_t* p = base + row * ld + kb;
    v16bf f;
#pragma unroll
    for (int i = 0; i < 8; ++i) f[i] = p[i];
#pragma unroll
    for (int i = 0; i < 8; ++i) f[i + 8] = p[16 + i];
    return f;
}

// B operand: 32(K) x 16(N) tile staged TRANSPOSED as Bt[n][k], row stride ld.
// Lane l holds column (l&15); K split 8-wide interleaved per half-wave.
__device__ __forceinline__ v16bf load_frag_b(const bf16_t* __restrict__ base, int ld) {
    int lane = threadIdx.x & 31;
    int col  = lane & 15;
    int kb   = (lane >> 4) << 3;
    const bf16_t* p = base + col * ld + kb;
    v16bf f;
#pragma unroll
    for (int i = 0; i < 8; ++i) f[i] = p[i];
#pragma unroll
    for (int i = 0; i < 8; ++i) f[i + 8] = p[16 + i];
    return f;
}

// ---------------------------------------------------------------------------
// Kernel: f32 -> packed bf16 elementwise (n even).
// ---------------------------------------------------------------------------
__global__ __launch_bounds__(256) void convert_bf16_kernel(
        const float* __restrict__ src, bf16_t* __restrict__ dst, int n2) {
    int i = blockIdx.x * 256 + threadIdx.x;
    if (i < n2) {
        float2 v = ((const float2*)src)[i];
        ((unsigned*)dst)[i] = pack2bf(v.x, v.y);
    }
}

// ---------------------------------------------------------------------------
// Kernel: W[K][N] f32 -> Bt[N][K] bf16 (tiled transpose + convert).
// ---------------------------------------------------------------------------
__global__ __launch_bounds__(256) void transpose_convert_kernel(
        const float* __restrict__ W, bf16_t* __restrict__ Bt, int K, int N) {
    __shared__ float tile[32][33];
    int kt = blockIdx.y * 32, nt = blockIdx.x * 32;
    int tx = threadIdx.x & 31, ty = threadIdx.x >> 5;     // 32 x 8
    for (int r = ty; r < 32; r += 8) {
        int k = kt + r, n = nt + tx;
        tile[r][tx] = (k < K && n < N) ? W[(size_t)k * N + n] : 0.0f;
    }
    __syncthreads();
    for (int i = threadIdx.x; i < 32 * 16; i += 256) {
        int rn = i >> 4, kp = (i & 15) * 2;
        int n = nt + rn, k = kt + kp;
        if (n < N && k < K)
            *(unsigned*)(Bt + (size_t)n * K + k) = pack2bf(tile[kp][rn], tile[kp + 1][rn]);
    }
}

// ---------------------------------------------------------------------------
// Kernel: axial RoPE cos/sin table. 196 positions x 32 pair-angles.
// ---------------------------------------------------------------------------
__global__ void rope_table_kernel(float* __restrict__ cosT, float* __restrict__ sinT) {
    for (int i = threadIdx.x; i < NPATCH * 32; i += blockDim.x) {
        int p = i >> 5, j = i & 31;
        int comp = j >> 4;
        int jj = j & 15;
        float freq  = powf(10.0f, -(float)jj / 16.0f);    // theta^{-4jj/64}
        float coord = comp ? (float)(p / 14) : (float)(p % 14);
        float ang = coord * freq;
        cosT[i] = cosf(ang);
        sinT[i] = sinf(ang);
    }
}

// ---------------------------------------------------------------------------
// Kernel: per-head Cayley matrix M = (I - S) * ((1+eps)I + S)^{-1}.
// Gauss-Jordan, no pivoting (A's symmetric part is (1+eps)I).
// ---------------------------------------------------------------------------
__global__ __launch_bounds__(256) void cayley_kernel(
        const float* __restrict__ s_raw, float* __restrict__ Mbuf) {
    __shared__ float S[64 * 65];
    __shared__ float A[64 * 65];
    __shared__ float Inv[64 * 65];
    __shared__ float pivinv;
    __shared__ float fcol[64];

    int h = blockIdx.x, t = threadIdx.x;
    const float* sr = s_raw + (size_t)h * 4096;

    for (int i = t; i < 4096; i += 256) {
        int r = i >> 6, c = i & 63;
        float sv = sr[r * 64 + c] - sr[c * 64 + r];
        S[r * 65 + c]   = sv;
        A[r * 65 + c]   = sv + (r == c ? 1.0f + EPSV : 0.0f);
        Inv[r * 65 + c] = (r == c) ? 1.0f : 0.0f;
    }
    __syncthreads();

    for (int c = 0; c < 64; ++c) {
        if (t == 0) pivinv = 1.0f / A[c * 65 + c];
        __syncthreads();
        float pv = pivinv;
        if (t < 64)       A[c * 65 + t] *= pv;
        else if (t < 128) Inv[c * 65 + (t - 64)] *= pv;
        else if (t < 192) { int r = t - 128; fcol[r] = (r == c) ? 0.0f : A[r * 65 + c]; }
        __syncthreads();
        int r = t >> 2, seg = t & 3;
        if (r != c) {
            float f = fcol[r];
#pragma unroll 4
            for (int q = 0; q < 32; ++q) {
                int j = seg * 32 + q;
                if (j < 64) A[r * 65 + j] -= f * A[c * 65 + j];
                else { int jj = j - 64; Inv[r * 65 + jj] -= f * Inv[c * 65 + jj]; }
            }
        }
        __syncthreads();
    }

    for (int rep = 0; rep < 16; ++rep) {
        int o = rep * 256 + t;
        int i = o >> 6, j = o & 63;
        float acc = Inv[i * 65 + j];
        for (int k = 0; k < 64; ++k) acc -= S[i * 65 + k] * Inv[k * 65 + j];
        Mbuf[(size_t)h * 4096 + i * 64 + j] = acc;
    }
}

// ---------------------------------------------------------------------------
// GEMM  C[M,N] = A[M,K] @ Bt[N,K]^T (+bias), bf16 in / fp32 out.
// 256 thr (8 waves), tile 128(M) x 64(N), K step 64, double-buffered LDS.
// All fragments preloaded before the 8 WMMAs -> graduated dscnt waits.
// ---------------------------------------------------------------------------
#define GBM 128
#define GBN 64
#define GBK 64
#define GLD (GBK + 8)

__global__ __launch_bounds__(256) void gemm_bf16_kernel(
        const bf16_t* __restrict__ Am, const bf16_t* __restrict__ Bt,
        const float* __restrict__ bias, float* __restrict__ C,
        int M, int N, int K) {
    __shared__ bf16_t As[2][GBM * GLD];
    __shared__ bf16_t Bs[2][GBN * GLD];

    int t = threadIdx.x;
    int wave = t >> 5, lane = t & 31;
    int bm = blockIdx.y * GBM, bn = blockIdx.x * GBN;

    const int arow = t >> 1, akoff = (t & 1) * 32;       // 128 rows x 64k, 32 bf16/thr
    const int gmA  = bm + arow;
    const int brow = t >> 2, bkoff = (t & 3) * 16;       // 64 rows x 64k, 16 bf16/thr
    const int gnB  = bn + brow;
    const bf16_t* aP = Am + (size_t)gmA * K + akoff;
    const bf16_t* bP = Bt + (size_t)gnB * K + bkoff;

    v8f acc[4];
#pragma unroll
    for (int i = 0; i < 4; ++i)
#pragma unroll
        for (int e = 0; e < 8; ++e) acc[i][e] = 0.0f;

    uint4 ar[4], br[2];
    const uint4 z4 = make_uint4(0u, 0u, 0u, 0u);

#define LOADREGS(kk)                                                          \
    do {                                                                      \
        if (gmA < M) {                                                        \
            const uint4* p = (const uint4*)(aP + (kk));                       \
            ar[0] = p[0]; ar[1] = p[1]; ar[2] = p[2]; ar[3] = p[3];           \
        } else { ar[0] = z4; ar[1] = z4; ar[2] = z4; ar[3] = z4; }            \
        if (gnB < N) {                                                        \
            const uint4* p = (const uint4*)(bP + (kk));                       \
            br[0] = p[0]; br[1] = p[1];                                       \
        } else { br[0] = z4; br[1] = z4; }                                    \
    } while (0)

#define STORELDS(buf)                                                         \
    do {                                                                      \
        uint4* d = (uint4*)&As[buf][arow * GLD + akoff];                      \
        d[0] = ar[0]; d[1] = ar[1]; d[2] = ar[2]; d[3] = ar[3];               \
        uint4* e = (uint4*)&Bs[buf][brow * GLD + bkoff];                      \
        e[0] = br[0]; e[1] = br[1];                                           \
    } while (0)

    const int nk = K / GBK;
    LOADREGS(0);
    STORELDS(0);
    __syncthreads();

    for (int i = 0; i < nk; ++i) {
        int cur = i & 1;
        if (i + 1 < nk) LOADREGS((i + 1) * GBK);

        // preload all fragments, then run 8 back-to-back WMMAs
        const bf16_t* abase = &As[cur][wave * 16 * GLD];
        v16bf af0 = load_frag_a(abase, GLD);
        v16bf af1 = load_frag_a(abase + 32, GLD);
        v16bf bfr[8];
#pragma unroll
        for (int ns = 0; ns < 4; ++ns) {
            const bf16_t* bbase = &Bs[cur][ns * 16 * GLD];
            bfr[ns * 2]     = load_frag_b(bbase, GLD);
            bfr[ns * 2 + 1] = load_frag_b(bbase + 32, GLD);
        }
#pragma unroll
        for (int ns = 0; ns < 4; ++ns) {
            acc[ns] = wmma_bf16(af0, bfr[ns * 2],     acc[ns]);
            acc[ns] = wmma_bf16(af1, bfr[ns * 2 + 1], acc[ns]);
        }

        if (i + 1 < nk) STORELDS(cur ^ 1);
        __syncthreads();
    }
#undef LOADREGS
#undef STORELDS

    int colL = lane & 15, rhalf = (lane >> 4) * 8;
#pragma unroll
    for (int ns = 0; ns < 4; ++ns) {
        int gn = bn + ns * 16 + colL;
        float bv = bias ? bias[gn] : 0.0f;
#pragma unroll
        for (int e = 0; e < 8; ++e) {
            int gm = bm + wave * 16 + rhalf + e;
            if (gm < M && gn < N) C[(size_t)gm * N + gn] = acc[ns][e] + bv;
        }
    }
}

// ---------------------------------------------------------------------------
// Fused RoPE + Cayley matvec on q/k patch-token rows (fp32, in place).
// Grid (2, 12, 64) = (q|k, head, batch). Block 256: 8 tokens per pass.
// ---------------------------------------------------------------------------
__global__ __launch_bounds__(256) void rope_cayley_kernel(
        float* __restrict__ qkv, const float* __restrict__ Mbuf,
        const float* __restrict__ cosT, const float* __restrict__ sinT) {
    __shared__ float Ml[64 * 65];
    __shared__ float xr[8 * 64];

    int qk = blockIdx.x, h = blockIdx.y, b = blockIdx.z;
    int t = threadIdx.x;
    const float* Mh = Mbuf + (size_t)h * 4096;
    for (int i = t; i < 4096; i += 256) Ml[(i >> 6) * 65 + (i & 63)] = Mh[i];
    __syncthreads();

    int basecol = qk * HIDDEN + h * HDIM;

    for (int p0 = 0; p0 < NPATCH; p0 += 8) {
        int g = t >> 5, j = t & 31;
        int p = p0 + g;
        if (p < NPATCH) {
            size_t row = (size_t)(b * SEQ + 1 + p) * QKVCOLS + basecol;
            float x0 = qkv[row + 2 * j], x1 = qkv[row + 2 * j + 1];
            float c = cosT[p * 32 + j], s = sinT[p * 32 + j];
            xr[g * 64 + 2 * j]     = x0 * c - x1 * s;
            xr[g * 64 + 2 * j + 1] = x0 * s + x1 * c;
        }
        __syncthreads();
#pragma unroll
        for (int rep = 0; rep < 2; ++rep) {
            int o = rep * 256 + t;
            int gg = o >> 6, d = o & 63;
            int pp = p0 + gg;
            if (pp < NPATCH) {
                float acc = 0.0f;
#pragma unroll 8
                for (int e = 0; e < 64; ++e) acc += Ml[d * 65 + e] * xr[gg * 64 + e];
                qkv[(size_t)(b * SEQ + 1 + pp) * QKVCOLS + basecol + d] = acc;
            }
        }
        __syncthreads();
    }
}

// ---------------------------------------------------------------------------
// Attention per (b,h). Keys padded to 224 (= 7*32). 32 q-rows per pass.
// Writes context as bf16 for the projection GEMM.
// Dynamic LDS: Kb[224][72] | Vt[64][232] | Qb[32][72] | Pb[32][232] bf16,
//              Sc[32][225] f32  = 110208 B.
// ---------------------------------------------------------------------------
#define NKP 224
#define KLD 72
#define VLD 232
#define SLD 225

__global__ __launch_bounds__(256) void attention_kernel(
        const float* __restrict__ qkv, bf16_t* __restrict__ ctx) {
    extern __shared__ char smem[];
    bf16_t* Kb = (bf16_t*)smem;
    bf16_t* Vt = Kb + NKP * KLD;
    bf16_t* Qb = Vt + 64 * VLD;
    bf16_t* Pb = Qb + 32 * KLD;
    float*  Sc = (float*)(Pb + 32 * VLD);

    int h = blockIdx.x, b = blockIdx.y;
    int t = threadIdx.x, wave = t >> 5, lane = t & 31;
    const float scale = 0.125f;

    // stage K rows (B^T for scores) + V transposed (B^T for out), bf16
    for (int i = t; i < NKP * (HDIM / 2); i += 256) {
        int n = i >> 5, d = (i & 31) * 2;
        float2 kv = make_float2(0.0f, 0.0f), vv = make_float2(0.0f, 0.0f);
        if (n < SEQ) {
            const float2* rb = (const float2*)(qkv + (size_t)(b * SEQ + n) * QKVCOLS + h * HDIM);
            kv = rb[(HIDDEN + d) >> 1];
            vv = rb[(2 * HIDDEN + d) >> 1];
        }
        *(unsigned*)(Kb + n * KLD + d) = pack2bf(kv.x, kv.y);
        Vt[d * VLD + n]       = f2bf(vv.x);
        Vt[(d + 1) * VLD + n] = f2bf(vv.y);
    }
    __syncthreads();

    for (int qt = 0; qt < 7; ++qt) {
        int qbase = qt * 32;
        for (int i = t; i < 32 * (HDIM / 2); i += 256) {
            int r = i >> 5, d = (i & 31) * 2;
            int n = qbase + r;
            float2 qv = make_float2(0.0f, 0.0f);
            if (n < SEQ)
                qv = ((const float2*)(qkv + (size_t)(b * SEQ + n) * QKVCOLS + h * HDIM))[d >> 1];
            *(unsigned*)(Qb + r * KLD + d) = pack2bf(qv.x, qv.y);
        }
        __syncthreads();

        // scores: 2 M-tiles x 14 N-tiles = 28 jobs over 8 waves
        for (int job = wave; job < 28; job += 8) {
            int mt = job / 14, nt = job % 14;
            v8f acc;
#pragma unroll
            for (int e = 0; e < 8; ++e) acc[e] = 0.0f;
            // preload both K-halves of A and B fragments, then 2 WMMAs
            v16bf a0  = load_frag_a(Qb + mt * 16 * KLD, KLD);
            v16bf a1  = load_frag_a(Qb + mt * 16 * KLD + 32, KLD);
            v16bf bb0 = load_frag_b(Kb + nt * 16 * KLD, KLD);
            v16bf bb1 = load_frag_b(Kb + nt * 16 * KLD + 32, KLD);
            acc = wmma_bf16(a0, bb0, acc);
            acc = wmma_bf16(a1, bb1, acc);
            int col = nt * 16 + (lane & 15);
            int rh  = (lane >> 4) * 8;
#pragma unroll
            for (int e = 0; e < 8; ++e) {
                int row = mt * 16 + rh + e;
                float v = acc[e] * scale;
                if (col >= SEQ) v = -1e30f;
                Sc[row * SLD + col] = v;
            }
        }
        __syncthreads();

        // softmax: wave handles 4 rows, lane-parallel over 224 cols
        for (int rr = 0; rr < 4; ++rr) {
            int row = wave * 4 + rr;
            float mx = -1e30f;
            for (int c = lane; c < NKP; c += 32) mx = fmaxf(mx, Sc[row * SLD + c]);
#pragma unroll
            for (int off = 16; off; off >>= 1) mx = fmaxf(mx, __shfl_xor(mx, off, 32));
            float sum = 0.0f;
            float ev[7];
#pragma unroll
            for (int ci = 0; ci < 7; ++ci) {
                int c = lane + ci * 32;
                float e = __expf(Sc[row * SLD + c] - mx);
                ev[ci] = e;
                sum += e;
            }
#pragma unroll
            for (int off = 16; off; off >>= 1) sum += __shfl_xor(sum, off, 32);
            float rinv = 1.0f / sum;
#pragma unroll
            for (int ci = 0; ci < 7; ++ci) {
                int c = lane + ci * 32;
                Pb[row * VLD + c] = f2bf(ev[ci] * rinv);
            }
        }
        __syncthreads();

        // out = P(32x224) @ V(224x64): one 16x16 job per wave
        {
            int mt = wave >> 2, nt = wave & 3;
            v8f acc;
#pragma unroll
            for (int e = 0; e < 8; ++e) acc[e] = 0.0f;
#pragma unroll
            for (int kk = 0; kk < NKP; kk += 32) {
                v16bf a  = load_frag_a(Pb + mt * 16 * VLD + kk, VLD);
                v16bf bb = load_frag_b(Vt + nt * 16 * VLD + kk, VLD);
                acc = wmma_bf16(a, bb, acc);
            }
            int d  = nt * 16 + (lane & 15);
            int rh = (lane >> 4) * 8;
#pragma unroll
            for (int e = 0; e < 8; ++e) {
                int r = mt * 16 + rh + e;
                int n = qbase + r;
                if (n < SEQ) ctx[(size_t)(b * SEQ + n) * HIDDEN + h * HDIM + d] = f2bf(acc[e]);
            }
        }
        __syncthreads();
    }
}

// ---------------------------------------------------------------------------
// Host launcher
// ---------------------------------------------------------------------------
extern "C" void kernel_launch(void* const* d_in, const int* in_sizes, int n_in,
                              void* d_out, int out_size, void* d_ws, size_t ws_size,
                              hipStream_t stream) {
    (void)in_sizes; (void)n_in; (void)out_size; (void)ws_size;

    const float* x      = (const float*)d_in[0];   // (64,197,768)
    const float* qkv_w  = (const float*)d_in[1];   // (768,2304)
    const float* proj_w = (const float*)d_in[2];   // (768,768)
    const float* proj_b = (const float*)d_in[3];   // (768,)
    const float* s_raw  = (const float*)d_in[4];   // (12,64,64)
    float* out = (float*)d_out;                    // (64,197,768)

    char* ws = (char*)d_ws;
    size_t off = 0;
    auto alloc = [&](size_t bytes) {
        size_t o = off;
        off = (off + bytes + 255) & ~(size_t)255;
        return o;
    };

    float*  qkvbuf = (float*) (ws + alloc((size_t)BNTOK * QKVCOLS * 4));
    bf16_t* xb     = (bf16_t*)(ws + alloc((size_t)BNTOK * HIDDEN * 2));
    bf16_t* wqkvT  = (bf16_t*)(ws + alloc((size_t)QKVCOLS * HIDDEN * 2));
    bf16_t* wprojT = (bf16_t*)(ws + alloc((size_t)HIDDEN * HIDDEN * 2));
    bf16_t* ctxb   = (bf16_t*)(ws + alloc((size_t)BNTOK * HIDDEN * 2));
    float*  Mbuf   = (float*) (ws + alloc((size_t)NHEADS * 4096 * 4));
    float*  cosT   = (float*) (ws + alloc((size_t)NPATCH * 32 * 4));
    float*  sinT   = (float*) (ws + alloc((size_t)NPATCH * 32 * 4));

    // 1. Tables + Cayley matrices
    rope_table_kernel<<<1, 256, 0, stream>>>(cosT, sinT);
    cayley_kernel<<<NHEADS, 256, 0, stream>>>(s_raw, Mbuf);

    // 2. One-time bf16 conversions
    {
        int n2 = (BNTOK * HIDDEN) / 2;
        convert_bf16_kernel<<<(n2 + 255) / 256, 256, 0, stream>>>(x, xb, n2);
        transpose_convert_kernel<<<dim3(QKVCOLS / 32, HIDDEN / 32), 256, 0, stream>>>(
            qkv_w, wqkvT, HIDDEN, QKVCOLS);
        transpose_convert_kernel<<<dim3(HIDDEN / 32, HIDDEN / 32), 256, 0, stream>>>(
            proj_w, wprojT, HIDDEN, HIDDEN);
    }

    // 3. QKV GEMM: (12608 x 768) @ (768 x 2304) -> fp32
    gemm_bf16_kernel<<<dim3(QKVCOLS / GBN, (BNTOK + GBM - 1) / GBM), 256, 0, stream>>>(
        xb, wqkvT, nullptr, qkvbuf, BNTOK, QKVCOLS, HIDDEN);

    // 4. RoPE + Cayley on q/k patch tokens (fp32, in place)
    rope_cayley_kernel<<<dim3(2, NHEADS, BATCH), 256, 0, stream>>>(qkvbuf, Mbuf, cosT, sinT);

    // 5. Attention per (b,h), writes bf16 context
    size_t attn_lds = (size_t)(NKP * KLD + 64 * VLD + 32 * KLD + 32 * VLD) * sizeof(bf16_t)
                    + (size_t)(32 * SLD) * sizeof(float);   // 110208 B
    attention_kernel<<<dim3(NHEADS, BATCH), 256, attn_lds, stream>>>(qkvbuf, ctxb);

    // 6. Output projection: (12608 x 768) @ (768 x 768) + bias -> d_out
    gemm_bf16_kernel<<<dim3(HIDDEN / GBN, (BNTOK + GBM - 1) / GBM), 256, 0, stream>>>(
        ctxb, wprojT, proj_b, out, BNTOK, HIDDEN, HIDDEN);
}
